// AttentionTSP_46875273068966
// MI455X (gfx1250) — matrix operand order repaired and checked by reading, not compiled, not run on hardware
//
#include <hip/hip_runtime.h>

// ---------------------------------------------------------------------------
// AttentionTSP forward for MI455X (gfx1250).
// Encoder + all projections run on v_wmma_f32_16x16x32_bf16 (bf16 in / f32
// accumulate).  All pure bf16 global->LDS staging copies use the CDNA5 async
// path (global_load_async_to_lds_b128 + s_wait_asynccnt).  Sequential pointer
// decoder runs as one persistent workgroup per batch instance with glimpse
// K/V staged in LDS.
// ---------------------------------------------------------------------------

constexpr int NB = 256;   // batch
constexpr int NS = 256;   // sequence length
constexpr int NE = 128;   // embedding dim
constexpr int NH = 128;   // hidden dim
constexpr int NHEADS = 4;
constexpr int NDH = 32;   // head dim
constexpr int NFF = 512;  // FFN dim
constexpr int NBS = NB * NS;

typedef __attribute__((ext_vector_type(16))) __bf16 v16bf;
typedef __attribute__((ext_vector_type(8)))  float  v8f;

// ---------------------------- small helpers -------------------------------

__device__ __forceinline__ float bf2f(unsigned short u) {
  union { unsigned int i; float f; } c; c.i = ((unsigned int)u) << 16; return c.f;
}
__device__ __forceinline__ unsigned short f2bf(float f) {
  union { float f; unsigned int i; } c; c.f = f;
  unsigned int x = c.i;
  return (unsigned short)((x + 0x7FFFu + ((x >> 16) & 1u)) >> 16);
}
__device__ __forceinline__ __bf16 bfbits(unsigned short u) {
  union { unsigned short s; __bf16 b; } c; c.s = u; return c.b;
}
__device__ __forceinline__ v8f vzero() {
  v8f z;
#pragma unroll
  for (int i = 0; i < 8; ++i) z[i] = 0.0f;
  return z;
}
__device__ __forceinline__ v8f wmma_bf16(v16bf a, v16bf b, v8f c) {
  // (neg_a, A, neg_b, B, c_mod, C, reuse_a, reuse_b)
  return __builtin_amdgcn_wmma_f32_16x16x32_bf16(false, a, false, b,
                                                 (short)0, c, false, false);
}

// CDNA5 async global->LDS 16-byte copy (ASYNCcnt tracked).  The LDS byte
// offset is the low 32 bits of the generic pointer (LDS aperture maps
// addr[31:0] onto the wave's LDS base).
__device__ __forceinline__ void async_copy_b128(void* sptr, const void* gptr) {
#if defined(__gfx1250__) && defined(__HIP_DEVICE_COMPILE__)
  unsigned int lo = (unsigned int)(unsigned long long)sptr;
  unsigned long long ga = (unsigned long long)gptr;
  asm volatile("global_load_async_to_lds_b128 %0, %1, off"
               :: "v"(lo), "v"(ga) : "memory");
#else
  const unsigned int* g = (const unsigned int*)gptr;
  unsigned int* s = (unsigned int*)sptr;
  s[0] = g[0]; s[1] = g[1]; s[2] = g[2]; s[3] = g[3];
#endif
}
__device__ __forceinline__ void async_wait_all() {
#if defined(__gfx1250__) && defined(__HIP_DEVICE_COMPILE__)
  asm volatile("s_wait_asynccnt 0x0" ::: "memory");
#endif
}

// A fragment (16x32 bf16) from row-major LDS tile, stride `ld` elements.
// Lanes 0-15 M=0..15; VGPR j<4 -> K base 0, j>=4 -> +16; lanes 16-31 add +8.
__device__ __forceinline__ v16bf loadA_u16(const unsigned short* X, int row0,
                                           int k0, int ld) {
  int lane = threadIdx.x & 31;
  int hi = (lane >> 4) & 1;
  const unsigned short* rp = X + (row0 + (lane & 15)) * ld + k0;
  v16bf a;
#pragma unroll
  for (int j = 0; j < 8; ++j) {
    int kb = ((j & 4) << 2) + hi * 8 + (j & 3) * 2;
    unsigned int p = *(const unsigned int*)(rp + kb);
    a[2 * j]     = bfbits((unsigned short)(p & 0xFFFFu));
    a[2 * j + 1] = bfbits((unsigned short)(p >> 16));
  }
  return a;
}

// Same A fragment but sourced from an f32 row-major LDS tile (converted).
__device__ __forceinline__ v16bf loadA_f32(const float* X, int row0,
                                           int k0, int ld) {
  int lane = threadIdx.x & 31;
  int hi = (lane >> 4) & 1;
  const float* rp = X + (row0 + (lane & 15)) * ld + k0;
  v16bf a;
#pragma unroll
  for (int j = 0; j < 8; ++j) {
    int kb = ((j & 4) << 2) + hi * 8 + (j & 3) * 2;
    a[2 * j]     = bfbits(f2bf(rp[kb]));
    a[2 * j + 1] = bfbits(f2bf(rp[kb + 1]));
  }
  return a;
}

// B fragment (32x16 bf16) from an N-major tile Wt[N][K] (stride ld elements).
// Lane = N, K = (j>=4)*16 + hiLane*8 + 2(j&3) + h  -> pairs contiguous in K.
__device__ __forceinline__ v16bf loadB_u16(const unsigned short* Wt, int n0,
                                           int k0, int ld) {
  int lane = threadIdx.x & 31;
  int hi = (lane >> 4) & 1;
  const unsigned short* rp = Wt + (n0 + (lane & 15)) * ld + k0;
  v16bf b;
#pragma unroll
  for (int j = 0; j < 8; ++j) {
    unsigned int p = *(const unsigned int*)(rp + ((j & 4) << 2) + hi * 8 + (j & 3) * 2);
    b[2 * j]     = bfbits((unsigned short)(p & 0xFFFFu));
    b[2 * j + 1] = bfbits((unsigned short)(p >> 16));
  }
  return b;
}

// ------------------------------ prep kernels -------------------------------

// W [K][N] f32  ->  Wt [N][K] bf16
__global__ __launch_bounds__(256) void transpose_bf16_kernel(
    const float* __restrict__ W, unsigned short* __restrict__ Wt,
    int K, int N) {
  int idx = blockIdx.x * 256 + threadIdx.x;
  if (idx >= N * K) return;
  int n = idx / K, k = idx - n * K;
  Wt[idx] = f2bf(W[(size_t)k * N + n]);
}

// h[r][j] = inputs[r][0]*embed[0][j] + inputs[r][1]*embed[1][j]
__global__ __launch_bounds__(256) void embed_kernel(
    const float* __restrict__ inp, const float* __restrict__ emb,
    float* __restrict__ h) {
  int idx = blockIdx.x * 256 + threadIdx.x;
  int r = idx >> 7, j = idx & 127;
  h[idx] = inp[r * 2] * emb[j] + inp[r * 2 + 1] * emb[NE + j];
}

// ------------------------------- WMMA GEMM ---------------------------------
// Y[M][N] = X[M][K] * W[K][N] + bias (+res) (relu?), X f32 or bf16,
// W pre-transposed bf16 Wt[N][K].  Tile 128x64, 8 waves of 32x32.
__global__ __launch_bounds__(256) void gemm_wmma_kernel(
    const float* __restrict__ Xf, const unsigned short* __restrict__ Xh,
    const unsigned short* __restrict__ Wt, const float* __restrict__ bias,
    const float* __restrict__ res, float* __restrict__ outF,
    unsigned short* __restrict__ outH, int M, int N, int K, int relu) {
  __shared__ __align__(16) unsigned short Xs[128 * 32];
  __shared__ __align__(16) unsigned short Ws[64 * 32];
  int t = threadIdx.x;
  int lane = t & 31, w = t >> 5;
  int m0 = blockIdx.x * 128;
  int n0 = blockIdx.y * 64;
  int wm = (w >> 1) * 32;
  int wn = (w & 1) * 32;
  v8f acc00 = vzero(), acc01 = vzero(), acc10 = vzero(), acc11 = vzero();
  for (int k0 = 0; k0 < K; k0 += 32) {
    {  // weight tile 64x32 bf16: one async b128 per thread
      int n = t >> 2, c = (t & 3) * 8;
      async_copy_b128(&Ws[n * 32 + c], Wt + (size_t)(n0 + n) * K + k0 + c);
    }
    if (Xf) {  // f32 activations: convert while staging (no async possible)
#pragma unroll
      for (int i = 0; i < 16; ++i) {
        int e = t + i * 256;
        int r = e >> 5, c = e & 31;
        Xs[e] = f2bf(Xf[(size_t)(m0 + r) * K + k0 + c]);
      }
    } else {  // bf16 activations: async copy 128x32 tile
#pragma unroll
      for (int i = 0; i < 2; ++i) {
        int e8 = (t + i * 256) * 8;
        int r = e8 >> 5, c = e8 & 31;
        async_copy_b128(&Xs[e8], Xh + (size_t)(m0 + r) * K + k0 + c);
      }
    }
    if (k0 + 32 < K) {  // global_prefetch_b8 for the next K tile
      const char* nxt = Xf ? (const char*)(Xf + (size_t)(m0 + (t >> 1)) * K + k0 + 32)
                           : (const char*)(Xh + (size_t)(m0 + (t >> 1)) * K + k0 + 32);
      __builtin_prefetch(nxt, 0, 0);
    }
    async_wait_all();
    __syncthreads();
    v16bf a0 = loadA_u16(Xs, wm, 0, 32);
    v16bf a1 = loadA_u16(Xs, wm + 16, 0, 32);
    v16bf b0 = loadB_u16(Ws, wn, 0, 32);
    v16bf b1 = loadB_u16(Ws, wn + 16, 0, 32);
    acc00 = wmma_bf16(a0, b0, acc00);
    acc01 = wmma_bf16(a0, b1, acc01);
    acc10 = wmma_bf16(a1, b0, acc10);
    acc11 = wmma_bf16(a1, b1, acc11);
    __syncthreads();
  }
  int hi8 = ((lane >> 4) & 1) * 8;
  int cn = lane & 15;
#pragma unroll
  for (int mi = 0; mi < 2; ++mi) {
#pragma unroll
    for (int ni = 0; ni < 2; ++ni) {
      v8f acc = mi ? (ni ? acc11 : acc10) : (ni ? acc01 : acc00);
#pragma unroll
      for (int r = 0; r < 8; ++r) {
        int m = m0 + wm + mi * 16 + r + hi8;
        int n = n0 + wn + ni * 16 + cn;
        float v = acc[r] + bias[n];
        if (relu) v = fmaxf(v, 0.0f);
        if (res) v += res[(size_t)m * N + n];
        if (outF) outF[(size_t)m * N + n] = v;
        if (outH) outH[(size_t)m * N + n] = f2bf(v);
      }
    }
  }
}

// -------------------------- encoder self-attention -------------------------
// One workgroup per (b, head).  K [S][32] and V^T [32][S] bf16 in LDS;
// scores via one WMMA K-step (DH=32); softmax in LDS; P*V via 8 WMMA steps.
__global__ __launch_bounds__(256) void attn_kernel(
    const unsigned short* __restrict__ Qb, const unsigned short* __restrict__ Kb,
    const unsigned short* __restrict__ Vb, unsigned short* __restrict__ Ob) {
  __shared__ __align__(16) unsigned short Ks[NS * NDH];   // [s][d]
  __shared__ __align__(16) unsigned short Vts[NDH * NS];  // [d][s]
  __shared__ __align__(16) unsigned short Qs[64 * NDH];
  __shared__ float Ss[64 * NS];
  int bh = blockIdx.x;
  int b = bh >> 2, h = bh & 3;
  int t = threadIdx.x, w = t >> 5, lane = t & 31;
  size_t rb = (size_t)b * NS;
#pragma unroll
  for (int i = 0; i < 2; ++i) {  // K tile: async 16B per thread
    int e8 = (t + i * 256) * 8;
    int s = e8 >> 5, d = e8 & 31;
    async_copy_b128(&Ks[e8], Kb + (rb + s) * NH + h * NDH + d);
  }
#pragma unroll 4
  for (int i = 0; i < 32; ++i) {  // V tile is transposed: manual stores
    int e = t + i * 256;
    int s = e >> 5, d = e & 31;
    Vts[d * NS + s] = Vb[(rb + s) * NH + h * NDH + d];
  }
  async_wait_all();
  __syncthreads();
  const float scale = 0.17677669529663687f;  // 1/sqrt(32)
  for (int qb = 0; qb < 4; ++qb) {
    int q0 = qb * 64;
    {  // Q tile 64x32: async 16B per thread
      int e8 = t * 8;
      int r = e8 >> 5, d = e8 & 31;
      async_copy_b128(&Qs[e8], Qb + (rb + q0 + r) * NH + h * NDH + d);
    }
    async_wait_all();
    __syncthreads();
    {  // scores: wave covers 16 rows x 128 cols
      int mf = w >> 1, half = w & 1;
      v16bf a = loadA_u16(Qs, mf * 16, 0, NDH);
      int hi8 = ((lane >> 4) & 1) * 8, cn = lane & 15;
#pragma unroll
      for (int nf = 0; nf < 8; ++nf) {
        int n0 = half * 128 + nf * 16;
        v16bf bm = loadB_u16(Ks, n0, 0, NDH);
        v8f c = vzero();
        c = wmma_bf16(a, bm, c);
#pragma unroll
        for (int r = 0; r < 8; ++r)
          Ss[(mf * 16 + r + hi8) * NS + n0 + cn] = c[r] * scale;
      }
    }
    __syncthreads();
    if (t < 64) {  // row softmax
      float mx = -3.0e38f;
      for (int s = 0; s < NS; ++s) mx = fmaxf(mx, Ss[t * NS + s]);
      float sm = 0.0f;
      for (int s = 0; s < NS; ++s) {
        float e = expf(Ss[t * NS + s] - mx);
        Ss[t * NS + s] = e;
        sm += e;
      }
      float inv = 1.0f / sm;
      for (int s = 0; s < NS; ++s) Ss[t * NS + s] *= inv;
    }
    __syncthreads();
    {  // P * V : wave computes 16 rows x 16 cols of the 64x32 output
      int mf = w >> 1, nf = w & 1;
      int n0 = nf * 16;
      v8f c = vzero();
#pragma unroll
      for (int ks = 0; ks < 8; ++ks) {
        v16bf a = loadA_f32(Ss, mf * 16, ks * 32, NS);
        v16bf bm = loadB_u16(Vts, n0, ks * 32, NS);
        c = wmma_bf16(a, bm, c);
      }
      int hi8 = ((lane >> 4) & 1) * 8, cn = lane & 15;
#pragma unroll
      for (int r = 0; r < 8; ++r) {
        int m = q0 + mf * 16 + r + hi8;
        Ob[(rb + m) * NH + h * NDH + n0 + cn] = f2bf(c[r]);
      }
    }
    __syncthreads();
  }
}

// ------------------------------ context prep -------------------------------

__global__ __launch_bounds__(128) void context_kernel(
    const float* __restrict__ enc, const float* __restrict__ Wh,
    const float* __restrict__ bh, float* __restrict__ hctx) {
  __shared__ float mean[NE];
  int b = blockIdx.x, t = threadIdx.x;
  float acc = 0.0f;
  for (int s = 0; s < NS; ++s) acc += enc[((size_t)b * NS + s) * NE + t];
  mean[t] = acc * (1.0f / NS);
  __syncthreads();
  float o = bh[t];
  for (int d = 0; d < NE; ++d) o += mean[d] * Wh[d * NE + t];
  hctx[b * NE + t] = o;
}

__global__ __launch_bounds__(128) void vinit_kernel(
    const float* __restrict__ initw, const float* __restrict__ Wv,
    const float* __restrict__ bv, float* __restrict__ vinit) {
  int t = threadIdx.x;
  float o = bv[t];
  for (int d = 0; d < 2 * NE; ++d) o += initw[d] * Wv[d * NE + t];
  vinit[t] = o;
}

// ------------------------------- decoder -----------------------------------
// One workgroup (256 threads) per batch instance; 256 sequential steps.
__global__ __launch_bounds__(256) void decoder_kernel(
    const unsigned short* __restrict__ Kg, const unsigned short* __restrict__ Vg,
    const unsigned short* __restrict__ Kp, const float* __restrict__ enc,
    const float* __restrict__ hctx, const float* __restrict__ vinit,
    const unsigned short* __restrict__ WqgT, const float* __restrict__ bqg,
    const unsigned short* __restrict__ WogT, const float* __restrict__ bog,
    const unsigned short* __restrict__ WqpT, const float* __restrict__ bqp,
    const unsigned short* __restrict__ VembT, const float* __restrict__ bve,
    const float* __restrict__ rand_u, float* __restrict__ out) {
  __shared__ __align__(16) unsigned short sK[NS * NH];  // glimpse K
  __shared__ __align__(16) unsigned short sV[NS * NH];  // glimpse V
  __shared__ float sHctx[NE], sQuery[NE], sFirst[NE];
  __shared__ float sQg[NH], sG[NH], sGlq[NE], sPq[NH];
  __shared__ float sAtt[NHEADS * NS];
  __shared__ float sLogit[NS], sProbs[NS], sCum[NS], sRed[NS];
  __shared__ unsigned int sMask[8];
  __shared__ float sFirstFlag;
  __shared__ int sAction;
  int b = blockIdx.x, t = threadIdx.x;
  size_t kvBase = (size_t)b * NS * NH;
#pragma unroll 4
  for (int i = 0; i < 16; ++i) {  // stage K/V: 2x64KB via async b128
    int e8 = (t + i * 256) * 8;
    async_copy_b128(&sK[e8], Kg + kvBase + e8);
    async_copy_b128(&sV[e8], Vg + kvBase + e8);
  }
  if (t < NE) {
    sHctx[t] = hctx[b * NE + t];
    sQuery[t] = sHctx[t] + vinit[t];
    sFirst[t] = 0.0f;
  }
  if (t < 8) sMask[t] = 0u;
  if (t == 0) sFirstFlag = 0.0f;
  async_wait_all();
  __syncthreads();
  const float sc_g = 0.17677669529663687f;  // 1/sqrt(32)
  const float sc_p = 0.08838834764831845f;  // 1/sqrt(128)
  for (int step = 0; step < NS; ++step) {
    float u_t = rand_u[step * NB + b];
    // glimpse query projection
    if (t < NH) {
      float acc = bqg[t];
      const unsigned short* wr = WqgT + (size_t)t * NE;
      for (int d = 0; d < NE; ++d) acc += sQuery[d] * bf2f(wr[d]);
      sQg[t] = acc;
    }
    __syncthreads();
    // glimpse scores (masked)
    for (int idx = t; idx < NHEADS * NS; idx += 256) {
      int h = idx >> 8, s = idx & 255;
      float v;
      if ((sMask[s >> 5] >> (s & 31)) & 1u) {
        v = -1.0e8f;
      } else {
        float acc = 0.0f;
        const unsigned short* kr = sK + s * NH + h * NDH;
        const float* qh = sQg + h * NDH;
        for (int d = 0; d < NDH; ++d) acc += qh[d] * bf2f(kr[d]);
        v = acc * sc_g;
      }
      sAtt[idx] = v;
    }
    __syncthreads();
    // per-head softmax (4 segments of 64 threads)
    {
      int h = t >> 6, l = t & 63;
      float pm = -3.0e38f;
      for (int c = 0; c < 4; ++c) pm = fmaxf(pm, sAtt[h * 256 + l * 4 + c]);
      sRed[t] = pm;
      __syncthreads();
      for (int stp = 32; stp > 0; stp >>= 1) {
        if (l < stp) sRed[t] = fmaxf(sRed[t], sRed[t + stp]);
        __syncthreads();
      }
      float mx = sRed[h * 64];
      __syncthreads();
      float ps = 0.0f;
      for (int c = 0; c < 4; ++c) {
        int s = l * 4 + c;
        float e = expf(sAtt[h * 256 + s] - mx);
        sAtt[h * 256 + s] = e;
        ps += e;
      }
      sRed[t] = ps;
      __syncthreads();
      for (int stp = 32; stp > 0; stp >>= 1) {
        if (l < stp) sRed[t] += sRed[t + stp];
        __syncthreads();
      }
      float inv = 1.0f / sRed[h * 64];
      __syncthreads();
      for (int c = 0; c < 4; ++c) sAtt[h * 256 + l * 4 + c] *= inv;
    }
    __syncthreads();
    // glimpse attention output
    if (t < NH) {
      int h = t >> 5, d = t & 31;
      float acc = 0.0f;
      for (int s = 0; s < NS; ++s)
        acc += sAtt[h * 256 + s] * bf2f(sV[s * NH + h * NDH + d]);
      sG[t] = acc;
    }
    __syncthreads();
    if (t < NE) {  // Wo
      float acc = bog[t];
      const unsigned short* wr = WogT + (size_t)t * NH;
      for (int k = 0; k < NH; ++k) acc += sG[k] * bf2f(wr[k]);
      sGlq[t] = acc;
    }
    __syncthreads();
    if (t < NH) {  // pointer query projection
      float acc = bqp[t];
      const unsigned short* wr = WqpT + (size_t)t * NE;
      for (int d = 0; d < NE; ++d) acc += sGlq[d] * bf2f(wr[d]);
      sPq[t] = acc;
    }
    __syncthreads();
    // pointer logits (thread t handles node s=t)
    {
      int s = t;
      float v;
      if ((sMask[s >> 5] >> (s & 31)) & 1u) {
        v = -1.0e8f;
      } else {
        float acc = 0.0f;
        const unsigned short* kr = Kp + ((size_t)b * NS + s) * NH;
        for (int d = 0; d < NH; ++d) acc += sPq[d] * bf2f(kr[d]);
        v = 10.0f * tanhf(acc * sc_p);
      }
      sLogit[t] = v;
    }
    __syncthreads();
    // softmax over 256
    sRed[t] = sLogit[t];
    __syncthreads();
    for (int stp = 128; stp > 0; stp >>= 1) {
      if (t < stp) sRed[t] = fmaxf(sRed[t], sRed[t + stp]);
      __syncthreads();
    }
    float mx = sRed[0];
    __syncthreads();
    float e = expf(sLogit[t] - mx);
    sRed[t] = e;
    __syncthreads();
    for (int stp = 128; stp > 0; stp >>= 1) {
      if (t < stp) sRed[t] += sRed[t + stp];
      __syncthreads();
    }
    float inv = 1.0f / sRed[0];
    __syncthreads();
    float p = e * inv;
    sProbs[t] = p;
    sCum[t] = p;
    __syncthreads();
    // inclusive scan (Hillis-Steele)
    for (int off = 1; off < 256; off <<= 1) {
      float v2 = (t >= off) ? sCum[t - off] : 0.0f;
      __syncthreads();
      sCum[t] += v2;
      __syncthreads();
    }
    // action = clip(count(cum < u), 0, S-1)
    sRed[t] = (sCum[t] < u_t) ? 1.0f : 0.0f;
    __syncthreads();
    for (int stp = 128; stp > 0; stp >>= 1) {
      if (t < stp) sRed[t] += sRed[t + stp];
      __syncthreads();
    }
    if (t == 0) {
      int a = (int)(sRed[0] + 0.5f);
      if (a > NS - 1) a = NS - 1;
      if (a < 0) a = 0;
      sAction = a;
      out[(size_t)b * NS + step] = logf(sProbs[a] + 1e-10f);
      out[(size_t)NB * NS + (size_t)b * NS + step] = (float)a;
      sMask[a >> 5] |= 1u << (a & 31);
    }
    __syncthreads();
    int action = sAction;
    float isf = (sFirstFlag == 0.0f) ? 1.0f : 0.0f;
    if (t < NE) {
      float se = enc[((size_t)b * NS + action) * NE + t];
      sFirst[t] = (isf > 0.5f) ? se : sFirst[t];
      sG[t] = se;  // reuse sG as selected node embedding
    }
    __syncthreads();
    if (t == 0) sFirstFlag = 1.0f;
    if (t < NE) {  // new_query = h_ctx + v_embed([first, sel]) + b
      float acc = bve[t];
      const unsigned short* wr = VembT + (size_t)t * (2 * NE);
      for (int k = 0; k < NE; ++k) acc += sFirst[k] * bf2f(wr[k]);
      for (int k = 0; k < NE; ++k) acc += sG[k] * bf2f(wr[NE + k]);
      sQuery[t] = sHctx[t] + acc;
    }
    __syncthreads();
  }
}

// ------------------------------- host side ---------------------------------

extern "C" void kernel_launch(void* const* d_in, const int* in_sizes, int n_in,
                              void* d_out, int out_size, void* d_ws,
                              size_t ws_size, hipStream_t stream) {
  (void)in_sizes; (void)n_in; (void)out_size; (void)ws_size;
  // Input order: jax-style sorted-key flatten of setup_inputs():
  // 0 inputs, 1 embed, [2..37] enc x3 (Wk.W,Wk.b,Wo.W,Wo.b,Wq.W,Wq.b,Wv.W,
  // Wv.b,ff1.W,ff1.b,ff2.W,ff2.b), [38..45] glimpse (Wk,Wo,Wq,Wv),
  // [46,47] h_ctx, 48 init_w, [49..52] pointer (Wk,Wq), [53,54] v_embed,
  // 55 rand_u.
  auto fin = [&](int i) { return (const float*)d_in[i]; };

  char* w = (char*)d_ws;
  size_t off = 0;
  auto galloc = [&](size_t n) -> void* {
    void* p = w + off;
    off += (n + 255) & ~(size_t)255;
    return p;
  };
  float* h0 = (float*)galloc((size_t)NBS * NE * 4);
  float* h1 = (float*)galloc((size_t)NBS * NE * 4);
  char* uni = (char*)galloc((size_t)NBS * NFF * 2);  // 67 MB union region
  unsigned short* Qb = (unsigned short*)uni;
  unsigned short* Kb = (unsigned short*)(uni + (size_t)NBS * NH * 2);
  unsigned short* Vb = (unsigned short*)(uni + (size_t)NBS * NH * 4);
  unsigned short* Ob = (unsigned short*)(uni + (size_t)NBS * NH * 6);
  unsigned short* FFb = (unsigned short*)uni;   // overlaps dead QKV+O
  unsigned short* Kgb = Qb;                     // overlap dead FF after encoder
  unsigned short* Vgb = Kb;
  unsigned short* Kpb = Vb;
  float* hctx = (float*)galloc((size_t)NB * NE * 4);
  float* vinitb = (float*)galloc(NE * 4);

  auto talloc = [&](int K, int N) -> unsigned short* {
    return (unsigned short*)galloc((size_t)K * N * 2);
  };
  auto tlaunch = [&](const float* Wsrc, unsigned short* Wdst, int K, int N) {
    int tot = K * N;
    transpose_bf16_kernel<<<(tot + 255) / 256, 256, 0, stream>>>(Wsrc, Wdst, K, N);
  };

  unsigned short *WkT[3], *WoT[3], *WqT[3], *WvT[3], *F1T[3], *F2T[3];
  const float *bk[3], *bo[3], *bq[3], *bv[3], *b1[3], *b2[3];
  for (int l = 0; l < 3; ++l) {
    int base = 2 + l * 12;
    WkT[l] = talloc(NE, NH);  tlaunch(fin(base + 0),  WkT[l], NE, NH);  bk[l] = fin(base + 1);
    WoT[l] = talloc(NH, NE);  tlaunch(fin(base + 2),  WoT[l], NH, NE);  bo[l] = fin(base + 3);
    WqT[l] = talloc(NE, NH);  tlaunch(fin(base + 4),  WqT[l], NE, NH);  bq[l] = fin(base + 5);
    WvT[l] = talloc(NE, NH);  tlaunch(fin(base + 6),  WvT[l], NE, NH);  bv[l] = fin(base + 7);
    F1T[l] = talloc(NE, NFF); tlaunch(fin(base + 8),  F1T[l], NE, NFF); b1[l] = fin(base + 9);
    F2T[l] = talloc(NFF, NE); tlaunch(fin(base + 10), F2T[l], NFF, NE); b2[l] = fin(base + 11);
  }
  unsigned short* WkgT = talloc(NE, NH); tlaunch(fin(38), WkgT, NE, NH);
  const float* bkg = fin(39);
  unsigned short* WogT = talloc(NH, NE); tlaunch(fin(40), WogT, NH, NE);
  const float* bog = fin(41);
  unsigned short* WqgT = talloc(NE, NH); tlaunch(fin(42), WqgT, NE, NH);
  const float* bqg = fin(43);
  unsigned short* WvgT = talloc(NE, NH); tlaunch(fin(44), WvgT, NE, NH);
  const float* bvg = fin(45);
  const float* Wh = fin(46);
  const float* bh = fin(47);
  const float* initw = fin(48);
  unsigned short* WkpT = talloc(NE, NH); tlaunch(fin(49), WkpT, NE, NH);
  const float* bkp = fin(50);
  unsigned short* WqpT = talloc(NE, NH); tlaunch(fin(51), WqpT, NE, NH);
  const float* bqp = fin(52);
  unsigned short* VembT = talloc(2 * NE, NE); tlaunch(fin(53), VembT, 2 * NE, NE);
  const float* bve = fin(54);
  const float* randu = fin(55);

  auto gemm = [&](const float* Xf, const unsigned short* Xh,
                  const unsigned short* Wt, const float* bias, const float* res,
                  float* outF, unsigned short* outH, int N, int K, int relu) {
    dim3 g(NBS / 128, N / 64);
    gemm_wmma_kernel<<<g, 256, 0, stream>>>(Xf, Xh, Wt, bias, res, outF, outH,
                                            NBS, N, K, relu);
  };

  // graph embedding
  embed_kernel<<<(NBS * NE) / 256, 256, 0, stream>>>(fin(0), fin(1), h0);

  // encoder
  for (int l = 0; l < 3; ++l) {
    gemm(h0, nullptr, WqT[l], bq[l], nullptr, nullptr, Qb, NH, NE, 0);
    gemm(h0, nullptr, WkT[l], bk[l], nullptr, nullptr, Kb, NH, NE, 0);
    gemm(h0, nullptr, WvT[l], bv[l], nullptr, nullptr, Vb, NH, NE, 0);
    attn_kernel<<<NB * NHEADS, 256, 0, stream>>>(Qb, Kb, Vb, Ob);
    gemm(nullptr, Ob, WoT[l], bo[l], h0, h1, nullptr, NE, NH, 0);
    gemm(h1, nullptr, F1T[l], b1[l], nullptr, nullptr, FFb, NFF, NE, 1);
    gemm(nullptr, FFb, F2T[l], b2[l], h1, h0, nullptr, NE, NFF, 0);
  }

  // decoder precompute: glimpse K/V and pointer K over encoded h0
  gemm(h0, nullptr, WkgT, bkg, nullptr, nullptr, Kgb, NH, NE, 0);
  gemm(h0, nullptr, WvgT, bvg, nullptr, nullptr, Vgb, NH, NE, 0);
  gemm(h0, nullptr, WkpT, bkp, nullptr, nullptr, Kpb, NH, NE, 0);
  context_kernel<<<NB, 128, 0, stream>>>(h0, Wh, bh, hctx);
  vinit_kernel<<<1, 128, 0, stream>>>(initw, fin(53), bve, vinitb);

  decoder_kernel<<<NB, 256, 0, stream>>>(Kgb, Vgb, Kpb, h0, hctx, vinitb,
                                         WqgT, bqg, WogT, bog, WqpT, bqp,
                                         VembT, bve, randu, (float*)d_out);
}
